// AttentionBlock_11132555231317
// MI455X (gfx1250) — matrix operand order repaired and checked by reading, not compiled, not run on hardware
//
#include <hip/hip_runtime.h>
#include <cmath>

typedef __attribute__((ext_vector_type(16))) _Float16 v16h;
typedef __attribute__((ext_vector_type(8)))  _Float16 v8h;
typedef __attribute__((ext_vector_type(8)))  float    v8f;
typedef __attribute__((ext_vector_type(4)))  int      i4v;

#define B_N     8
#define HWN     4096          // 64*64 tokens per batch
#define C_N     64
#define ROWS    (B_N * HWN)   // 32768
#define GROUPS_ 8
#define CPG     (C_N / GROUPS_)   // 8
#define EPS_GN  1e-3f
#define SCALE_Q 0.125f        // 64^-0.5
#define KT      64            // keys per attention tile

#ifndef __has_builtin
#define __has_builtin(x) 0
#endif
#if __has_builtin(__builtin_amdgcn_global_load_async_to_lds_b128) && \
    __has_builtin(__builtin_amdgcn_s_wait_asynccnt)
#define HAVE_ASYNC_LDS 1
#endif

#ifdef HAVE_ASYNC_LDS
typedef __attribute__((address_space(1))) i4v gbl_i4;
typedef __attribute__((address_space(3))) i4v lds_i4;
__device__ __forceinline__ void async_cp16(const void* g, const void* l) {
  __builtin_amdgcn_global_load_async_to_lds_b128(
      (gbl_i4*)(size_t)g, (lds_i4*)(unsigned)(size_t)l, 0, 0);
}
#endif

__device__ __forceinline__ v8f wmma_f16(v16h a, v16h b, v8f c) {
  return __builtin_amdgcn_wmma_f32_16x16x32_f16(false, a, false, b, (short)0, c,
                                                false, false);
}

// A-fragment (16x32 f16) from a row-major [*,64] f16 matrix.
// lane: m = lane&15 ; element j -> k = kb+j, kb = 16*(lane>>4)
__device__ __forceinline__ v16h load_a_rowmajor(const _Float16* __restrict__ base,
                                                int row0, int cbase, int lane) {
  int m  = lane & 15;
  int kb = (lane >> 4) * 16;
  const _Float16* p = base + (size_t)(row0 + m) * C_N + cbase + kb;
  v8h lo = *(const v8h*)(p);
  v8h hi = *(const v8h*)(p + 8);
  v16h a;
  #pragma unroll
  for (int i = 0; i < 8; ++i) { a[i] = lo[i]; a[8 + i] = hi[i]; }
  return a;
}

// A-fragment (16x32 f16) from an LDS row-major tile with row stride `ld` halves.
__device__ __forceinline__ v16h load_a_lds(const _Float16* base, int ld, int kbase,
                                           int lane) {
  int m  = lane & 15;
  int kb = (lane >> 4) * 16;
  const _Float16* p = base + m * ld + kbase + kb;
  v8h lo = *(const v8h*)(p);
  v8h hi = *(const v8h*)(p + 8);
  v16h a;
  #pragma unroll
  for (int i = 0; i < 8; ++i) { a[i] = lo[i]; a[8 + i] = hi[i]; }
  return a;
}

// ---------------------------------------------------------------- group norm
__global__ void gn_kernel(const float* __restrict__ x,
                          const float* __restrict__ gamma,
                          const float* __restrict__ beta,
                          float* __restrict__ hf32,
                          _Float16* __restrict__ hf16) {
  __shared__ float s_sum[256], s_sq[256];
  int bg = blockIdx.x;            // B*GROUPS = 64 blocks
  int b = bg >> 3, g = bg & 7;
  int tid = threadIdx.x;
  const int N = HWN * CPG;        // 32768 elements per (b,g)
  float sum = 0.f, sq = 0.f;
  for (int i = tid; i < N; i += 256) {
    int hw = i >> 3, cc = i & 7;
    float v = x[(size_t)(b * HWN + hw) * C_N + g * CPG + cc];
    sum += v; sq += v * v;
  }
  s_sum[tid] = sum; s_sq[tid] = sq;
  __syncthreads();
  for (int s = 128; s > 0; s >>= 1) {
    if (tid < s) { s_sum[tid] += s_sum[tid + s]; s_sq[tid] += s_sq[tid + s]; }
    __syncthreads();
  }
  float mean = s_sum[0] / (float)N;
  float var  = s_sq[0] / (float)N - mean * mean;
  float rstd = rsqrtf(var + EPS_GN);
  for (int i = tid; i < N; i += 256) {
    int hw = i >> 3, cc = i & 7;
    int c = g * CPG + cc;
    size_t off = (size_t)(b * HWN + hw) * C_N + c;
    float v = (x[off] - mean) * rstd * gamma[c] + beta[c];
    hf32[off] = v;
    hf16[off] = (_Float16)v;
  }
}

// ------------------------------------------------------------ QKV projection
// wT in LDS is transposed: wT[n*64 + c] so B-fragment loads are contiguous.
__device__ __forceinline__ v16h load_b_wT(const _Float16* wT, int ntile, int ctile,
                                          int lane) {
  int n  = ntile * 16 + (lane & 15);
  int cb = ctile * 32 + (lane >> 4) * 16;
  return *(const v16h*)(wT + n * 64 + cb);
}

__global__ void qkv_kernel(const _Float16* __restrict__ h16,
                           const float* __restrict__ wq, const float* __restrict__ bq,
                           const float* __restrict__ wk, const float* __restrict__ bk,
                           const float* __restrict__ wv, const float* __restrict__ bv,
                           _Float16* __restrict__ q, _Float16* __restrict__ k,
                           _Float16* __restrict__ vT) {
  __shared__ __align__(32) _Float16 swq[64 * 64], swk[64 * 64], swv[64 * 64];
  int tid = threadIdx.x;
  for (int i = tid; i < 4096; i += 256) {        // transpose-stage weights
    int c = i >> 6, n = i & 63;
    swq[n * 64 + c] = (_Float16)wq[i];
    swk[n * 64 + c] = (_Float16)wk[i];
    swv[n * 64 + c] = (_Float16)wv[i];
  }
  __syncthreads();

  int lane = tid & 31, wave = tid >> 5;
  int row0 = blockIdx.x * 128 + wave * 16;       // 8 waves * 16 rows
  int n15 = lane & 15, hi = lane >> 4;

  v16h a0 = load_a_rowmajor(h16, row0, 0, lane);
  v16h a1 = load_a_rowmajor(h16, row0, 32, lane);

  #pragma unroll
  for (int nt = 0; nt < 4; ++nt) {
    int n = nt * 16 + n15;
    // ---- Q (scaled by C^-0.5)
    v8f acc; float bb = bq[n];
    #pragma unroll
    for (int r = 0; r < 8; ++r) acc[r] = bb;
    acc = wmma_f16(a0, load_b_wT(swq, nt, 0, lane), acc);
    acc = wmma_f16(a1, load_b_wT(swq, nt, 1, lane), acc);
    #pragma unroll
    for (int r = 0; r < 8; ++r)
      q[(size_t)(row0 + hi * 8 + r) * C_N + n] = (_Float16)(acc[r] * SCALE_Q);
    // ---- K
    bb = bk[n];
    #pragma unroll
    for (int r = 0; r < 8; ++r) acc[r] = bb;
    acc = wmma_f16(a0, load_b_wT(swk, nt, 0, lane), acc);
    acc = wmma_f16(a1, load_b_wT(swk, nt, 1, lane), acc);
    #pragma unroll
    for (int r = 0; r < 8; ++r)
      k[(size_t)(row0 + hi * 8 + r) * C_N + n] = (_Float16)acc[r];
    // ---- V, stored transposed: vT[(b*64 + n)*HWN + t]
    bb = bv[n];
    #pragma unroll
    for (int r = 0; r < 8; ++r) acc[r] = bb;
    acc = wmma_f16(a0, load_b_wT(swv, nt, 0, lane), acc);
    acc = wmma_f16(a1, load_b_wT(swv, nt, 1, lane), acc);
    int b = row0 >> 12;                 // row0 / 4096
    int t = (row0 & 4095) + hi * 8;
    v8h vv;
    #pragma unroll
    for (int r = 0; r < 8; ++r) vv[r] = (_Float16)acc[r];
    *(v8h*)(vT + (size_t)(b * 64 + n) * HWN + t) = vv;   // contiguous 16B store
  }
}

// ------------------------------------------------------- fused flash attention
// 128 threads = 4 waves; each wave owns 16 query rows; 64-key tiles.
__global__ void attn_kernel(const _Float16* __restrict__ q,
                            const _Float16* __restrict__ k,
                            const _Float16* __restrict__ vT,
                            _Float16* __restrict__ po) {
  __shared__ __align__(32) _Float16 sk[KT * 64];       // K tile  [t][c]     8KB
  __shared__ __align__(32) _Float16 sv[64 * KT];       // V tile  [c][t]     8KB
  __shared__ __align__(32) float    sS[4][16 * KT];    // per-wave scores   16KB
  __shared__ __align__(32) _Float16 sP[4][16 * KT];    // per-wave probs     8KB
  __shared__ float sM[4][16], sL[4][16], sCorr[4][16];
  __shared__ float sRmax[4][32], sRsum[4][32];

  int tid  = threadIdx.x;
  int lane = tid & 31, wave = tid >> 5;
  int blk  = blockIdx.x;                // B * (HWN/64) = 512 blocks
  int b    = blk >> 6;
  int q0   = (blk & 63) * 64 + wave * 16;
  int grow = b * HWN + q0;
  int n15 = lane & 15, hi = lane >> 4;

  v16h aq0 = load_a_rowmajor(q, grow, 0, lane);
  v16h aq1 = load_a_rowmajor(q, grow, 32, lane);

  v8f o[4];
  #pragma unroll
  for (int nt = 0; nt < 4; ++nt) o[nt] = (v8f){};
  if (lane < 16) { sM[wave][lane] = -INFINITY; sL[wave][lane] = 0.f; }

  // per-thread staging coordinates: 128 threads move 64B each per tile
  int st  = tid >> 1;                  // K row   (0..63)
  int sc0 = (tid & 1) * 32;            // K col base
  int scv = tid >> 1;                  // V channel (0..63)
  int st0 = (tid & 1) * 32;            // V token base

  for (int kt = 0; kt < HWN / KT; ++kt) {
    int kb = kt * KT;
    __syncthreads();                               // previous-tile reads done
    {
      const _Float16* gk = &k[((size_t)(b * HWN + kb + st)) * C_N + sc0];
      const _Float16* gv = &vT[((size_t)(b * 64 + scv)) * HWN + kb + st0];
      _Float16* lk = &sk[st * 64 + sc0];
      _Float16* lv = &sv[scv * KT + st0];
#ifdef HAVE_ASYNC_LDS
      async_cp16(gk,      lk);
      async_cp16(gk + 8,  lk + 8);
      async_cp16(gk + 16, lk + 16);
      async_cp16(gk + 24, lk + 24);
      async_cp16(gv,      lv);
      async_cp16(gv + 8,  lv + 8);
      async_cp16(gv + 16, lv + 16);
      async_cp16(gv + 24, lv + 24);
      __builtin_amdgcn_s_wait_asynccnt(0);
#else
      *(v16h*)lk        = *(const v16h*)gk;
      *(v16h*)(lk + 16) = *(const v16h*)(gk + 16);
      *(v16h*)lv        = *(const v16h*)gv;
      *(v16h*)(lv + 16) = *(const v16h*)(gv + 16);
#endif
      // prefetch next tile one iteration ahead (wraps harmlessly on last iter)
      int kbn = ((kt + 1) & (HWN / KT - 1)) * KT;
      __builtin_prefetch(&k[((size_t)(b * HWN + kbn + st)) * C_N + sc0], 0, 1);
      __builtin_prefetch(&vT[((size_t)(b * 64 + scv)) * HWN + kbn + st0], 0, 1);
    }
    __syncthreads();

    // S = Q K^T : four 16x16 output tiles over this 64-key strip
    #pragma unroll
    for (int nt = 0; nt < 4; ++nt) {
      v8f s = (v8f){};
      v16h b0 = *(const v16h*)&sk[(nt * 16 + n15) * 64 + hi * 16];
      v16h b1 = *(const v16h*)&sk[(nt * 16 + n15) * 64 + 32 + hi * 16];
      s = wmma_f16(aq0, b0, s);
      s = wmma_f16(aq1, b1, s);
      #pragma unroll
      for (int r = 0; r < 8; ++r)
        sS[wave][(hi * 8 + r) * KT + nt * 16 + n15] = s[r];
    }

    // online softmax, all 32 lanes: lane owns row n15, cols [hi*32, hi*32+32)
    {
      const float* srow = &sS[wave][n15 * KT + hi * 32];
      float mx = srow[0];
      #pragma unroll
      for (int j = 1; j < 32; ++j) mx = fmaxf(mx, srow[j]);
      sRmax[wave][lane] = mx;
      float tmax = fmaxf(mx, sRmax[wave][lane ^ 16]);   // wave-internal: in-order LDS
      float mOld = sM[wave][n15];
      float mNew = fmaxf(mOld, tmax);
      float corr = __expf(mOld - mNew);                 // 0 on first tile
      float lsum = 0.f;
      _Float16* prow = &sP[wave][n15 * KT + hi * 32];
      #pragma unroll
      for (int j = 0; j < 32; ++j) {
        float p = __expf(srow[j] - mNew);
        lsum += p;
        prow[j] = (_Float16)p;
      }
      sRsum[wave][lane] = lsum;
      float lNew = sL[wave][n15] * corr + lsum + sRsum[wave][lane ^ 16];
      sM[wave][n15] = mNew;        // both halves write identical values
      sL[wave][n15] = lNew;
      sCorr[wave][n15] = corr;
    }

    // rescale accumulators by per-row correction
    float f[8];
    #pragma unroll
    for (int r = 0; r < 8; ++r) f[r] = sCorr[wave][hi * 8 + r];
    #pragma unroll
    for (int nt = 0; nt < 4; ++nt)
      #pragma unroll
      for (int r = 0; r < 8; ++r) o[nt][r] *= f[r];

    // O += P V : two 32-key strips x four 16-channel tiles
    v16h pa0 = load_a_lds(&sP[wave][0], KT, 0, lane);
    v16h pa1 = load_a_lds(&sP[wave][0], KT, 32, lane);
    #pragma unroll
    for (int nt = 0; nt < 4; ++nt) {
      int c = nt * 16 + n15;
      o[nt] = wmma_f16(pa0, *(const v16h*)&sv[c * KT + hi * 16], o[nt]);
      o[nt] = wmma_f16(pa1, *(const v16h*)&sv[c * KT + 32 + hi * 16], o[nt]);
    }
  }

  float linv[8];
  #pragma unroll
  for (int r = 0; r < 8; ++r) linv[r] = 1.f / sL[wave][hi * 8 + r];
  #pragma unroll
  for (int nt = 0; nt < 4; ++nt) {
    int n = nt * 16 + n15;
    #pragma unroll
    for (int r = 0; r < 8; ++r)
      po[(size_t)(grow + hi * 8 + r) * C_N + n] = (_Float16)(o[nt][r] * linv[r]);
  }
}

// ------------------------------------------------- final projection + residual
__global__ void proj_kernel(const _Float16* __restrict__ po,
                            const float* __restrict__ wp, const float* __restrict__ bp,
                            const float* __restrict__ hf32,
                            float* __restrict__ out) {
  __shared__ __align__(32) _Float16 swp[64 * 64];
  int tid = threadIdx.x;
  for (int i = tid; i < 4096; i += 256) {
    int c = i >> 6, n = i & 63;
    swp[n * 64 + c] = (_Float16)wp[i];
  }
  __syncthreads();

  int lane = tid & 31, wave = tid >> 5;
  int row0 = blockIdx.x * 128 + wave * 16;
  int n15 = lane & 15, hi = lane >> 4;

  v16h a0 = load_a_rowmajor(po, row0, 0, lane);
  v16h a1 = load_a_rowmajor(po, row0, 32, lane);

  #pragma unroll
  for (int nt = 0; nt < 4; ++nt) {
    int n = nt * 16 + n15;
    v8f acc; float bb = bp[n];
    #pragma unroll
    for (int r = 0; r < 8; ++r) acc[r] = bb;
    acc = wmma_f16(a0, load_b_wT(swp, nt, 0, lane), acc);
    acc = wmma_f16(a1, load_b_wT(swp, nt, 1, lane), acc);
    #pragma unroll
    for (int r = 0; r < 8; ++r) {
      size_t off = (size_t)(row0 + hi * 8 + r) * C_N + n;
      out[off] = hf32[off] + acc[r];
    }
  }
}

// ---------------------------------------------------------------------- launch
extern "C" void kernel_launch(void* const* d_in, const int* in_sizes, int n_in,
                              void* d_out, int out_size, void* d_ws, size_t ws_size,
                              hipStream_t stream) {
  const float* x     = (const float*)d_in[0];
  const float* gamma = (const float*)d_in[1];
  const float* beta  = (const float*)d_in[2];
  const float* wq    = (const float*)d_in[3];
  const float* bq    = (const float*)d_in[4];
  const float* wk    = (const float*)d_in[5];
  const float* bk    = (const float*)d_in[6];
  const float* wv    = (const float*)d_in[7];
  const float* bv    = (const float*)d_in[8];
  const float* wp    = (const float*)d_in[9];
  const float* bp    = (const float*)d_in[10];
  float* out = (float*)d_out;

  char* ws = (char*)d_ws;
  float*    hf32 = (float*)ws;      ws += (size_t)ROWS * C_N * sizeof(float);
  _Float16* hf16 = (_Float16*)ws;   ws += (size_t)ROWS * C_N * sizeof(_Float16);
  _Float16* qb   = (_Float16*)ws;   ws += (size_t)ROWS * C_N * sizeof(_Float16);
  _Float16* kb   = (_Float16*)ws;   ws += (size_t)ROWS * C_N * sizeof(_Float16);
  _Float16* vT   = (_Float16*)ws;   ws += (size_t)ROWS * C_N * sizeof(_Float16);
  _Float16* po   = (_Float16*)ws;   ws += (size_t)ROWS * C_N * sizeof(_Float16);

  gn_kernel<<<B_N * GROUPS_, 256, 0, stream>>>(x, gamma, beta, hf32, hf16);
  qkv_kernel<<<ROWS / 128, 256, 0, stream>>>(hf16, wq, bq, wk, bk, wv, bv,
                                             qb, kb, vT);
  attn_kernel<<<B_N * (HWN / 64), 128, 0, stream>>>(qb, kb, vT, po);
  proj_kernel<<<ROWS / 128, 256, 0, stream>>>(po, wp, bp, hf32, out);
}
